// TMRModel_60275571032086
// MI455X (gfx1250) — matrix operand order repaired and checked by reading, not compiled, not run on hardware
//
#include <hip/hip_runtime.h>
#include <hip/hip_bf16.h>
#include <math.h>

// ---------------- problem constants (match reference) ----------------
#define B_ 32
#define T_ 512
#define D_ 768
#define S_ 64
#define V_ 30522
#define C_ 16
#define STEPS_ 4
#define DECAY_ 0.9f
#define SCORE_CLIP_ 20.0f
#define EPS_ 1e-9f
#define LN_EPS_ 1e-5f

typedef _Float16 half16 __attribute__((ext_vector_type(16)));
typedef float    float8 __attribute__((ext_vector_type(8)));

union HFrag {
    half16 h;
    unsigned int u[8];
};
union U4H8 {
    uint4 v;
    _Float16 h[8];
};
union U2H4 {
    uint2 v;
    _Float16 h[4];
};

// low 32 bits of a generic pointer into LDS == wave-relative LDS byte offset
__device__ __forceinline__ unsigned ldsOffset(const void* p) {
    return (unsigned)(unsigned long long)p;
}

// async memory->LDS copy of 16 bytes per lane (ASYNCcnt-tracked, no VGPR data)
__device__ __forceinline__ void asyncCopyB128(unsigned ldsAddr, const void* gptr) {
    asm volatile("global_load_async_to_lds_b128 %0, %1, off"
                 :: "v"(ldsAddr), "v"(gptr)
                 : "memory");
}
__device__ __forceinline__ void waitAsync0() {
    asm volatile("s_wait_asynccnt 0x0" ::: "memory");
}

// ---------------- small elementwise kernels (float4-vectorized) ----------------

// x[b,t,d] = tok_emb[ids[b,t], d] + pos_emb[t, d]   -> f16 (4 d at a time)
__global__ void embed_kernel(const int* __restrict__ ids,
                             const float4* __restrict__ tok,
                             const float4* __restrict__ pos,
                             uint2* __restrict__ xh, long n4) {
    const int D4 = D_ / 4;
    long i = (long)blockIdx.x * blockDim.x + threadIdx.x;
    long stride = (long)gridDim.x * blockDim.x;
    for (; i < n4; i += stride) {
        int d4  = (int)(i % D4);
        long bt = i / D4;
        int t   = (int)(bt % T_);
        int id  = ids[bt];
        float4 tv = tok[(long)id * D4 + d4];
        float4 pv = pos[(long)t * D4 + d4];
        U2H4 o;
        o.h[0] = (_Float16)(tv.x + pv.x);
        o.h[1] = (_Float16)(tv.y + pv.y);
        o.h[2] = (_Float16)(tv.z + pv.z);
        o.h[3] = (_Float16)(tv.w + pv.w);
        xh[i] = o.v;
    }
}

// valid[b] = sum_t (mask[b,t] > 0)
__global__ void valid_kernel(const int* __restrict__ mask, float* __restrict__ valid) {
    int b = threadIdx.x;
    if (b < B_) {
        int c = 0;
        for (int t = 0; t < T_; ++t) c += (mask[(long)b * T_ + t] > 0) ? 1 : 0;
        valid[b] = (float)c;
    }
}

// out[n][k] = (f16) in[k][n]   (LDS tile transpose; coalesced both sides)
__global__ __launch_bounds__(256) void transpose_f32_to_f16_kernel(
    const float* __restrict__ in, _Float16* __restrict__ out, int K, int N) {
    __shared__ __align__(16) _Float16 tile[64][72];
    const int tid = threadIdx.x;
    const int k0 = blockIdx.y * 64;
    const int n0 = blockIdx.x * 64;
    // load 64(k) x 64(n) floats as float4, convert, store transposed into LDS
#pragma unroll
    for (int i = 0; i < 4; ++i) {
        int e = tid + i * 256;
        int r = e >> 4;          // k row 0..63
        int c4 = e & 15;         // float4 chunk along n
        float4 v = *(const float4*)(in + (long)(k0 + r) * N + n0 + c4 * 4);
        tile[c4 * 4 + 0][r] = (_Float16)v.x;
        tile[c4 * 4 + 1][r] = (_Float16)v.y;
        tile[c4 * 4 + 2][r] = (_Float16)v.z;
        tile[c4 * 4 + 3][r] = (_Float16)v.w;
    }
    __syncthreads();
    // write 64(n) x 64(k) halves as b128
#pragma unroll
    for (int i = 0; i < 2; ++i) {
        int e = tid + i * 256;
        int nr = e >> 3;
        int kc = (e & 7) * 8;
        *(uint4*)(out + (long)(n0 + nr) * K + k0 + kc) = *(const uint4*)&tile[nr][kc];
    }
}

// M f32 = broadcast(mem_init), plus f16 shadow copy
__global__ void minit_kernel(const float4* __restrict__ mem,
                             float4* __restrict__ Mf, uint2* __restrict__ Mh, long n4) {
    const long sd4 = (long)S_ * D_ / 4;
    long i = (long)blockIdx.x * blockDim.x + threadIdx.x;
    long stride = (long)gridDim.x * blockDim.x;
    for (; i < n4; i += stride) {
        float4 v = mem[i % sd4];
        Mf[i] = v;
        U2H4 o;
        o.h[0] = (_Float16)v.x; o.h[1] = (_Float16)v.y;
        o.h[2] = (_Float16)v.z; o.h[3] = (_Float16)v.w;
        Mh[i] = o.v;
    }
}

// M = clip(DECAY*M + (1-DECAY)*write, -50, 50); refresh f16 shadow
__global__ void mupdate_kernel(float4* __restrict__ Mf, uint2* __restrict__ Mh,
                               const float4* __restrict__ wr, long n4) {
    long i = (long)blockIdx.x * blockDim.x + threadIdx.x;
    long stride = (long)gridDim.x * blockDim.x;
    for (; i < n4; i += stride) {
        float4 m = Mf[i];
        float4 w = wr[i];
        float4 v;
        v.x = fminf(fmaxf(DECAY_ * m.x + (1.0f - DECAY_) * w.x, -50.0f), 50.0f);
        v.y = fminf(fmaxf(DECAY_ * m.y + (1.0f - DECAY_) * w.y, -50.0f), 50.0f);
        v.z = fminf(fmaxf(DECAY_ * m.z + (1.0f - DECAY_) * w.z, -50.0f), 50.0f);
        v.w = fminf(fmaxf(DECAY_ * m.w + (1.0f - DECAY_) * w.w, -50.0f), 50.0f);
        Mf[i] = v;
        U2H4 o;
        o.h[0] = (_Float16)v.x; o.h[1] = (_Float16)v.y;
        o.h[2] = (_Float16)v.z; o.h[3] = (_Float16)v.w;
        Mh[i] = o.v;
    }
}

// ---------------- batched WMMA GEMM ----------------
// C[M,N] (+batch) = A[M,K] * B[K,N], f16 inputs, f32 accumulate.
// TRANS_A: A stored as [K,M] row-major (element A[m,k] = A[k*lda+m])
// TRANS_B: B stored as [N,K] row-major (element B[k,n] = B[n*ldb+k])
// Block tile 64(M) x 64(N); K-chunk 64; 8 waves (4Mx2N), each wave owns a
// 16x32 strip (2 accumulators, A-fragment reuse) -> 4 WMMAs per stage/wave.
// Double-buffered LDS, one barrier per stage. Direct-copy operands are
// staged with GLOBAL_LOAD_ASYNC_TO_LDS_B128 (ASYNCcnt): the wave issues the
// next-chunk copy and proceeds straight to its WMMAs; s_wait_asynccnt 0
// before the stage barrier publishes the LDS writes. Transpose-scatter
// operands (TRANS_A / non-TRANS_B) keep synchronous load+scatter. Epilogue
// stages C through LDS for coalesced b128 stores. Shapes are exact tile
// multiples (no predication).
template <int TRANS_A, int TRANS_B, int OUT_HALF>
__global__ __launch_bounds__(256, 1) void gemm_wmma_kernel(
    const _Float16* __restrict__ A, const _Float16* __restrict__ Bm,
    void* __restrict__ Cout, int M, int N, int K, int lda, int ldb, int ldc,
    long sA, long sB, long sC) {
    // 72-half row pitch = 144B: 16B aligned + bank skew
    __shared__ __align__(16) _Float16 lA[2][64][72];
    __shared__ __align__(16) _Float16 lB[2][64][72];  // [n][k]

    constexpr bool USE_ASYNC = (!TRANS_A) || TRANS_B;

    const int tid   = threadIdx.x;
    const int lane  = tid & 31;
    const int wave  = tid >> 5;
    const int waveM = wave >> 1;  // 0..3
    const int waveN = wave & 1;   // 0..1
    const int m0 = blockIdx.y * 64;
    const int n0 = blockIdx.x * 64;

    A  += (long)blockIdx.z * sA;
    Bm += (long)blockIdx.z * sB;

    float8 acc0 = {};
    float8 acc1 = {};

    // tile coordinates for this thread (uint4 = 8 halves along fastest dim)
    const int e0r = tid >> 3,         e0c = (tid & 7) * 8;
    const int e1r = (tid + 256) >> 3, e1c = ((tid + 256) & 7) * 8;

    // stage chunk at k-offset kc into LDS buffer buf
    auto stageTiles = [&](int kc, int buf) {
        if constexpr (!TRANS_A) {
            asyncCopyB128(ldsOffset(&lA[buf][e0r][e0c]),
                          A + (long)(m0 + e0r) * lda + kc + e0c);
            asyncCopyB128(ldsOffset(&lA[buf][e1r][e1c]),
                          A + (long)(m0 + e1r) * lda + kc + e1c);
        } else {
            U4H8 ra0, ra1;
            ra0.v = *(const uint4*)(A + (long)(kc + e0r) * lda + m0 + e0c);
            ra1.v = *(const uint4*)(A + (long)(kc + e1r) * lda + m0 + e1c);
#pragma unroll
            for (int j = 0; j < 8; ++j) lA[buf][e0c + j][e0r] = ra0.h[j];
#pragma unroll
            for (int j = 0; j < 8; ++j) lA[buf][e1c + j][e1r] = ra1.h[j];
        }
        if constexpr (TRANS_B) {
            asyncCopyB128(ldsOffset(&lB[buf][e0r][e0c]),
                          Bm + (long)(n0 + e0r) * ldb + kc + e0c);
            asyncCopyB128(ldsOffset(&lB[buf][e1r][e1c]),
                          Bm + (long)(n0 + e1r) * ldb + kc + e1c);
        } else {
            U4H8 rb0, rb1;
            rb0.v = *(const uint4*)(Bm + (long)(kc + e0r) * ldb + n0 + e0c);
            rb1.v = *(const uint4*)(Bm + (long)(kc + e1r) * ldb + n0 + e1c);
#pragma unroll
            for (int j = 0; j < 8; ++j) lB[buf][e0c + j][e0r] = rb0.h[j];
#pragma unroll
            for (int j = 0; j < 8; ++j) lB[buf][e1c + j][e1r] = rb1.h[j];
        }
    };

    const int nChunks = K >> 6;

    stageTiles(0, 0);
    if constexpr (USE_ASYNC) waitAsync0();
    __syncthreads();

    const int arow  = waveM * 16 + (lane & 15);
    const int kAo   = (lane >> 4) << 4;   // A frag: lanes 16-31 hold K+16
    const int bcol0 = waveN * 32 + (lane & 15);
    const int bcol1 = bcol0 + 16;
    const int kBo   = (lane >> 4) << 4;   // B frag: lanes 16-31 hold K+16

    for (int c = 0; c < nChunks; ++c) {
        const int cur = c & 1;
        const bool more = (c + 1) < nChunks;
        // stage next chunk into the idle buffer (safe: its last readers
        // finished before the barrier that ended iteration c-1)
        if (more) stageTiles((c + 1) << 6, cur ^ 1);

#pragma unroll
        for (int hk = 0; hk < 2; ++hk) {
            const int kk = hk * 32;
            HFrag a, b0, b1;
#pragma unroll
            for (int v = 0; v < 8; ++v) {
                int ka = kk + kAo + (v << 1);
                a.u[v] = *(const unsigned int*)&lA[cur][arow][ka];
                int kb = kk + kBo + (v << 1);
                b0.u[v] = *(const unsigned int*)&lB[cur][bcol0][kb];
                b1.u[v] = *(const unsigned int*)&lB[cur][bcol1][kb];
            }
            acc0 = __builtin_amdgcn_wmma_f32_16x16x32_f16(
                false, a.h, false, b0.h, (short)0, acc0, false, false);
            acc1 = __builtin_amdgcn_wmma_f32_16x16x32_f16(
                false, a.h, false, b1.h, (short)0, acc1, false, false);
        }

        if (more) {
            if constexpr (USE_ASYNC) waitAsync0();
            __syncthreads();
        }
    }

    // ---- epilogue: stage C tile in LDS (reuse lA), then coalesced b128 ----
    // C/D frag layout: lanes 0-15 rows r+0, lanes 16-31 rows r+8; col = lane&15
    __syncthreads();  // everyone done reading lA/lB
    const int rowL = waveM * 16 + ((lane >> 4) << 3);
    const int colL = waveN * 32 + (lane & 15);
    if constexpr (OUT_HALF) {
        _Float16* lC = (_Float16*)&lA[0][0][0];   // pitch 72 halves (9216B)
#pragma unroll
        for (int r = 0; r < 8; ++r) {
            lC[(rowL + r) * 72 + colL]      = (_Float16)acc0[r];
            lC[(rowL + r) * 72 + colL + 16] = (_Float16)acc1[r];
        }
        __syncthreads();
        _Float16* Cp = (_Float16*)Cout + (long)blockIdx.z * sC;
#pragma unroll
        for (int i = 0; i < 2; ++i) {
            int e = tid + i * 256;
            int r = e >> 3, c8 = (e & 7) * 8;
            *(uint4*)(Cp + (long)(m0 + r) * ldc + n0 + c8) =
                *(const uint4*)&lC[r * 72 + c8];
        }
    } else {
        float* lC = (float*)&lA[0][0][0];         // pitch 68 floats (17408B)
#pragma unroll
        for (int r = 0; r < 8; ++r) {
            lC[(rowL + r) * 68 + colL]      = acc0[r];
            lC[(rowL + r) * 68 + colL + 16] = acc1[r];
        }
        __syncthreads();
        float* Cp = (float*)Cout + (long)blockIdx.z * sC;
#pragma unroll
        for (int i = 0; i < 4; ++i) {
            int e = tid + i * 256;
            int r = e >> 4, c4 = (e & 15) * 4;
            *(uint4*)(Cp + (long)(m0 + r) * ldc + n0 + c4) =
                *(const uint4*)&lC[r * 68 + c4];
        }
    }
}

// ---------------- masked softmax over S=64; 8 rows/block, 1 wave/row ----------------
__global__ __launch_bounds__(256) void softmax_kernel(
    const float* __restrict__ scores, float* __restrict__ W,
    _Float16* __restrict__ Wh, const int* __restrict__ mask,
    const float* __restrict__ valid, float scale) {
    const int lane = threadIdx.x & 31;
    const int w    = threadIdx.x >> 5;
    const int bt   = blockIdx.x * 8 + w;
    const int b    = bt / T_;
    const int t    = bt - b * T_;
    const float* row = scores + (long)bt * S_;

    float x0 = row[lane];
    float x1 = row[lane + 32];
    x0 = fminf(fmaxf(x0 * scale, -SCORE_CLIP_), SCORE_CLIP_);
    x1 = fminf(fmaxf(x1 * scale, -SCORE_CLIP_), SCORE_CLIP_);

    const bool pad = (mask[(long)b * T_ + t] <= 0) && (valid[b] > 0.0f);

    float o0, o1;
    if (pad) {
        o0 = 0.0f; o1 = 0.0f;   // reference: masked row -> nan_to_num -> 0
    } else {
        float mx = fmaxf(x0, x1);
#pragma unroll
        for (int off = 16; off > 0; off >>= 1)
            mx = fmaxf(mx, __shfl_xor(mx, off, 32));
        float e0 = __expf(x0 - mx);
        float e1 = __expf(x1 - mx);
        float s = e0 + e1;
#pragma unroll
        for (int off = 16; off > 0; off >>= 1)
            s += __shfl_xor(s, off, 32);
        s = fmaxf(s, EPS_);
        o0 = e0 / s; o1 = e1 / s;
    }
    float* wr = W + (long)bt * S_;
    wr[lane]      = o0;
    wr[lane + 32] = o1;
    _Float16* wh = Wh + (long)bt * S_;
    wh[lane]      = (_Float16)o0;
    wh[lane + 32] = (_Float16)o1;
}

// colsum[b,s] = sum_t W[b,t,s]  (masked rows are already zero)
__global__ void colsum_kernel(const float* __restrict__ W, float* __restrict__ cs) {
    const int b = blockIdx.x;
    const int s = threadIdx.x;  // 64 threads -> 256B coalesced per step
    float acc = 0.0f;
    const float* base = W + (long)b * T_ * S_ + s;
    for (int t = 0; t < T_; ++t) acc += base[(long)t * S_];
    cs[(long)b * S_ + s] = acc;
}

// pooled = (colsum @ M)/valid ; LayerNorm ; logits = pooled @ cls_w + cls_b
__global__ __launch_bounds__(256) void head_kernel(
    const float* __restrict__ cs, const float* __restrict__ Mf,
    const float* __restrict__ valid, const float* __restrict__ ln_s,
    const float* __restrict__ ln_b, const float* __restrict__ cw,
    const float* __restrict__ cb, float* __restrict__ out) {
    __shared__ float pbuf[D_];
    __shared__ float red[256];
    const int b   = blockIdx.x;
    const int tid = threadIdx.x;
    const float inv = 1.0f / fmaxf(valid[b], 1.0f);

    for (int d = tid; d < D_; d += 256) {
        float acc = 0.0f;
#pragma unroll 8
        for (int s = 0; s < S_; ++s)
            acc += cs[(long)b * S_ + s] * Mf[((long)b * S_ + s) * D_ + d];
        pbuf[d] = acc * inv;
    }
    __syncthreads();

    // mean
    float part = 0.0f;
    for (int d = tid; d < D_; d += 256) part += pbuf[d];
    red[tid] = part;
    __syncthreads();
    for (int st = 128; st > 0; st >>= 1) {
        if (tid < st) red[tid] += red[tid + st];
        __syncthreads();
    }
    const float mu = red[0] / (float)D_;
    __syncthreads();

    // variance
    part = 0.0f;
    for (int d = tid; d < D_; d += 256) {
        float df = pbuf[d] - mu;
        part += df * df;
    }
    red[tid] = part;
    __syncthreads();
    for (int st = 128; st > 0; st >>= 1) {
        if (tid < st) red[tid] += red[tid + st];
        __syncthreads();
    }
    const float rstd = rsqrtf(red[0] / (float)D_ + LN_EPS_);
    __syncthreads();

    for (int d = tid; d < D_; d += 256)
        pbuf[d] = (pbuf[d] - mu) * rstd * ln_s[d] + ln_b[d];
    __syncthreads();

    for (int c = 0; c < C_; ++c) {
        float p = 0.0f;
        for (int d = tid; d < D_; d += 256) p += pbuf[d] * cw[(long)d * C_ + c];
        red[tid] = p;
        __syncthreads();
        for (int st = 128; st > 0; st >>= 1) {
            if (tid < st) red[tid] += red[tid + st];
            __syncthreads();
        }
        if (tid == 0) {
            float r = red[0] + cb[c];
            if (!isfinite(r)) r = 0.0f;
            out[(long)b * C_ + c] = r;
        }
        __syncthreads();
    }
}

// ---------------- host orchestration ----------------

extern "C" void kernel_launch(void* const* d_in, const int* in_sizes, int n_in,
                              void* d_out, int out_size, void* d_ws, size_t ws_size,
                              hipStream_t stream) {
    const int*   ids  = (const int*)d_in[0];
    const int*   mask = (const int*)d_in[1];
    const float* tok  = (const float*)d_in[2];
    const float* pos  = (const float*)d_in[3];
    const float* Wq   = (const float*)d_in[4];
    const float* Wk   = (const float*)d_in[5];
    const float* Wv   = (const float*)d_in[6];
    const float* mem  = (const float*)d_in[7];
    const float* lns  = (const float*)d_in[8];
    const float* lnb  = (const float*)d_in[9];
    const float* cw   = (const float*)d_in[10];
    const float* cb   = (const float*)d_in[11];
    float* out = (float*)d_out;

    char* wsp = (char*)d_ws;
    auto alloc = [&](size_t bytes) -> char* {
        char* p = wsp;
        wsp += (bytes + 255) & ~(size_t)255;
        return p;
    };

    const long nBTD = (long)B_ * T_ * D_;
    const long nBSD = (long)B_ * S_ * D_;
    const long nBTS = (long)B_ * T_ * S_;
    const long nDD  = (long)D_ * D_;

    _Float16* WqT   = (_Float16*)alloc(nDD * 2);   // [N][K] transposed f16 weights
    _Float16* WkT   = (_Float16*)alloc(nDD * 2);
    _Float16* WvT   = (_Float16*)alloc(nDD * 2);
    _Float16* xh    = (_Float16*)alloc(nBTD * 2);
    _Float16* Qh    = (_Float16*)alloc(nBTD * 2);
    _Float16* Vh    = (_Float16*)alloc(nBTD * 2);
    float*    Mf    = (float*)alloc(nBSD * 4);
    _Float16* Mh    = (_Float16*)alloc(nBSD * 2);
    _Float16* Kmh   = (_Float16*)alloc(nBSD * 2);
    float*    scW   = (float*)alloc(nBTS * 4);     // scores, reused as softmax W (f32)
    _Float16* Wh    = (_Float16*)alloc(nBTS * 2);
    float*    wrf   = (float*)alloc(nBSD * 4);
    float*    csum  = (float*)alloc((long)B_ * S_ * 4);
    float*    validf= (float*)alloc(B_ * 4);

    const float inv_sqrt_d = 1.0f / sqrtf((float)D_);

    // 1) transpose-convert weights to f16 [N][K] so all weight GEMMs hit the
    //    fully vectorized TRANS_B path
    dim3 tgrid(D_ / 64, D_ / 64);
    transpose_f32_to_f16_kernel<<<tgrid, 256, 0, stream>>>(Wq, WqT, D_, D_);
    transpose_f32_to_f16_kernel<<<tgrid, 256, 0, stream>>>(Wk, WkT, D_, D_);
    transpose_f32_to_f16_kernel<<<tgrid, 256, 0, stream>>>(Wv, WvT, D_, D_);

    // 2) embedding gather + positional add -> f16 x
    embed_kernel<<<dim3(2048), dim3(256), 0, stream>>>(
        ids, (const float4*)tok, (const float4*)pos, (uint2*)xh, nBTD / 4);

    // 3) mask valid counts
    valid_kernel<<<dim3(1), dim3(32), 0, stream>>>(mask, validf);

    // 4) Qt = x @ Wq, Vt = x @ Wv   (M=16384, N=768, K=768; B = W^T [N][K])
    gemm_wmma_kernel<0, 1, 1><<<dim3(D_ / 64, (B_ * T_) / 64, 1), 256, 0, stream>>>(
        xh, WqT, Qh, B_ * T_, D_, D_, D_, D_, D_, 0, 0, 0);
    gemm_wmma_kernel<0, 1, 1><<<dim3(D_ / 64, (B_ * T_) / 64, 1), 256, 0, stream>>>(
        xh, WvT, Vh, B_ * T_, D_, D_, D_, D_, D_, 0, 0, 0);

    // 5) init memory M
    minit_kernel<<<dim3(1024), dim3(256), 0, stream>>>(
        (const float4*)mem, (float4*)Mf, (uint2*)Mh, nBSD / 4);

    // 6) recurrence: 4 update steps + 1 final read step
    for (int step = 0; step <= STEPS_; ++step) {
        // Km = M @ Wk   (batched: M=64, N=768, K=768; B = Wk^T)
        gemm_wmma_kernel<0, 1, 1><<<dim3(D_ / 64, 1, B_), 256, 0, stream>>>(
            Mh, WkT, Kmh, S_, D_, D_, D_, D_, D_,
            (long)S_ * D_, 0, (long)S_ * D_);

        // scores = Qt @ Km^T  (batched: M=512, N=64, K=768; B stored [N,K])
        gemm_wmma_kernel<0, 1, 0><<<dim3(S_ / 64, T_ / 64, B_), 256, 0, stream>>>(
            Qh, Kmh, scW, T_, S_, D_, D_, D_, S_,
            (long)T_ * D_, (long)S_ * D_, (long)T_ * S_);

        // masked softmax (scale + clip inside); writes W f32 (in place) + f16
        softmax_kernel<<<dim3(B_ * T_ / 8), dim3(256), 0, stream>>>(
            scW, scW, Wh, mask, validf, inv_sqrt_d);

        if (step < STEPS_) {
            // write = W^T @ Vt  (batched: M=64, N=768, K=512; A stored [K,M])
            gemm_wmma_kernel<1, 0, 0><<<dim3(D_ / 64, 1, B_), 256, 0, stream>>>(
                Wh, Vh, wrf, S_, D_, T_, S_, D_, D_,
                (long)T_ * S_, (long)T_ * D_, (long)S_ * D_);

            // M = clip(decay*M + (1-decay)*write)
            mupdate_kernel<<<dim3(1024), dim3(256), 0, stream>>>(
                (float4*)Mf, (uint2*)Mh, (const float4*)wrf, nBSD / 4);
        }
    }

    // 7) pooled = (sum_t W) @ M / valid  -> LN -> logits
    colsum_kernel<<<dim3(B_), dim3(S_), 0, stream>>>(scW, csum);
    head_kernel<<<dim3(B_), dim3(256), 0, stream>>>(
        csum, Mf, validf, lns, lnb, cw, cb, out);
}
